// Critic_35974646071852
// MI455X (gfx1250) — compile-verified, matrix operand order checked by reading, not run on hardware
//
#include <hip/hip_runtime.h>
#include <hip/hip_bf16.h>
#include <stdint.h>

typedef __attribute__((ext_vector_type(16))) _Float16 v16h;
typedef __attribute__((ext_vector_type(8)))  float    v8f;

#define NEG_SLOPE 0.1f

// ---------------------------------------------------------------------------
// Simple fill kernel
// ---------------------------------------------------------------------------
__global__ void fill_kernel(float* __restrict__ p, float v, int n) {
    int i = blockIdx.x * blockDim.x + threadIdx.x;
    if (i < n) p[i] = v;
}

// deg accumulation: one thread per edge, atomic add 1.0 to deg[dst]
__global__ void deg_accum_kernel(const int* __restrict__ dst, float* __restrict__ deg, int E) {
    int e = blockIdx.x * blockDim.x + threadIdx.x;
    if (e < E) atomicAdd(&deg[dst[e]], 1.0f);
}

__global__ void rsqrt_kernel(float* __restrict__ p, int n) {
    int i = blockIdx.x * blockDim.x + threadIdx.x;
    if (i < n) p[i] = rsqrtf(p[i]);
}

// ---------------------------------------------------------------------------
// Pack W [IN_F, OUT_F] f32 row-major into per-lane B-fragment order, f16.
// Fragment (kc, nt): lane holds col = nt*16 + (lane&15); element j holds
// K = kc*32 + (lane>>4)*8 + (j<8 ? j : j+8). Stored contiguously so the GEMM
// reads one v16h (32B) per fragment per lane, fully coalesced.
// ---------------------------------------------------------------------------
__global__ void pack_w_kernel(const float* __restrict__ W, _Float16* __restrict__ Wf,
                              int OUT_F, int total) {
    int idx = blockIdx.x * blockDim.x + threadIdx.x;
    if (idx >= total) return;
    const int NT   = OUT_F / 16;
    const int j    = idx & 15;
    const int lane = (idx >> 4) & 31;
    const int t    = idx >> 9;          // kc*NT + nt
    const int kc   = t / NT;
    const int nt   = t - kc * NT;
    const int col  = nt * 16 + (lane & 15);
    const int k    = kc * 32 + (lane >> 4) * 8 + (j < 8 ? j : j + 8);
    Wf[idx] = (_Float16)W[k * OUT_F + col];
}

// ---------------------------------------------------------------------------
// WMMA GEMM: H[nrows, OUT_F] = X[nrows, IN_F] @ W[IN_F, OUT_F]
// One wave (32 threads) per 16-row tile. nrows must be a multiple of 16.
// A built on the fly from f32 X (float4 loads + cvt_pk), B from pre-packed
// f16 fragments (one 32B vector load each). f32 accumulate.
// ---------------------------------------------------------------------------
template <int IN_F, int OUT_F>
__global__ __launch_bounds__(32)
void gemm_wmma_kernel(const float* __restrict__ X, const v16h* __restrict__ Wf,
                      float* __restrict__ H) {
    constexpr int NT = OUT_F / 16;   // N tiles
    constexpr int KC = IN_F / 32;    // K chunks of 32
    const int tile = blockIdx.x;
    const int lane = threadIdx.x & 31;
    const int r    = lane & 15;      // row within tile (A) / col within ntile (B)
    const int kh   = lane >> 4;      // K-half selector
    const int row  = tile * 16 + r;

    v8f acc[NT] = {};

    #pragma unroll
    for (int kc = 0; kc < KC; ++kc) {
        // ---- A fragment: row-major 16x32 f16 slice of X ----
        const float* xp = X + (long long)row * IN_F + kc * 32 + kh * 8;
        const float4 a0 = *(const float4*)(xp + 0);
        const float4 a1 = *(const float4*)(xp + 4);
        const float4 a2 = *(const float4*)(xp + 16);
        const float4 a3 = *(const float4*)(xp + 20);
        v16h a;
        a[0]  = (_Float16)a0.x; a[1]  = (_Float16)a0.y; a[2]  = (_Float16)a0.z; a[3]  = (_Float16)a0.w;
        a[4]  = (_Float16)a1.x; a[5]  = (_Float16)a1.y; a[6]  = (_Float16)a1.z; a[7]  = (_Float16)a1.w;
        a[8]  = (_Float16)a2.x; a[9]  = (_Float16)a2.y; a[10] = (_Float16)a2.z; a[11] = (_Float16)a2.w;
        a[12] = (_Float16)a3.x; a[13] = (_Float16)a3.y; a[14] = (_Float16)a3.z; a[15] = (_Float16)a3.w;

        #pragma unroll
        for (int nt = 0; nt < NT; ++nt) {
            // ---- B fragment: one coalesced 32B vector load ----
            const v16h b = Wf[(kc * NT + nt) * 32 + lane];
            acc[nt] = __builtin_amdgcn_wmma_f32_16x16x32_f16(
                false, a, false, b, (short)0, acc[nt], false, false);
        }
    }

    // ---- store C: element i -> (M = i + kh*8, N = lane&15) ----
    #pragma unroll
    for (int nt = 0; nt < NT; ++nt) {
        #pragma unroll
        for (int i = 0; i < 8; ++i) {
            const int orow = tile * 16 + i + kh * 8;
            H[(long long)orow * OUT_F + nt * 16 + r] = acc[nt][i];
        }
    }
}

// ---------------------------------------------------------------------------
// agg[n,f] = h[n,f] * dis[n]^2  (self-loop contribution initialization)
// ---------------------------------------------------------------------------
__global__ void self_init_kernel(const float* __restrict__ h, const float* __restrict__ dis,
                                 float* __restrict__ agg, int n, int F) {
    int gid = blockIdx.x * blockDim.x + threadIdx.x;
    if (gid < n * F) {
        int node = gid / F;
        float d = dis[node];
        agg[gid] = h[gid] * d * d;
    }
}

// ---------------------------------------------------------------------------
// Edge scatter: agg[dst] += h[src] * dis[src]*dis[dst]
// One thread per (edge, float4 feature group)
// ---------------------------------------------------------------------------
template <int F>
__global__ void edge_accum_kernel(const float* __restrict__ h,
                                  const int* __restrict__ src, const int* __restrict__ dst,
                                  const float* __restrict__ dis,
                                  float* __restrict__ agg, int E) {
    constexpr int G = F / 4;
    int gid = blockIdx.x * blockDim.x + threadIdx.x;
    if (gid >= E * G) return;
    int e = gid / G;
    int g = gid - e * G;
    int s = src[e];
    int d = dst[e];
    float norm = dis[s] * dis[d];
    const float4 hv = ((const float4*)(h + (long long)s * F))[g];
    float* base = agg + (long long)d * F + g * 4;
    atomicAdd(base + 0, hv.x * norm);
    atomicAdd(base + 1, hv.y * norm);
    atomicAdd(base + 2, hv.z * norm);
    atomicAdd(base + 3, hv.w * norm);
}

// ---------------------------------------------------------------------------
// a[n,f] = (act ? lrelu : id)(a[n,f] + b[f])
// ---------------------------------------------------------------------------
__global__ void bias_act_kernel(float* __restrict__ a, const float* __restrict__ b,
                                int n, int F, int act) {
    int gid = blockIdx.x * blockDim.x + threadIdx.x;
    if (gid < n * F) {
        int f = gid % F;
        float v = a[gid] + b[f];
        if (act) v = v > 0.0f ? v : NEG_SLOPE * v;
        a[gid] = v;
    }
}

// ---------------------------------------------------------------------------
// Global max pool: g[batch[n], f] = max_n h[n, f]   (F = 64)
// ---------------------------------------------------------------------------
__device__ __forceinline__ void atomicMaxF(float* addr, float v) {
    if (v >= 0.0f) atomicMax((int*)addr, __float_as_int(v));
    else           atomicMin((unsigned int*)addr, __float_as_uint(v));
}

__global__ void pool_kernel(const float* __restrict__ h, const int* __restrict__ batch,
                            float* __restrict__ g, int n) {
    int gid = blockIdx.x * blockDim.x + threadIdx.x;
    if (gid >= n * 64) return;
    int node = gid >> 6;
    int f = gid & 63;
    atomicMaxF(&g[batch[node] * 64 + f], h[gid]);
}

// ---------------------------------------------------------------------------
// Final MLP on [64,64]: lrelu(g@Wl1+bl1) -> lrelu(@Wl2+bl2) -> @Wl3+bl3
// Single workgroup, LDS staging. Trivial FLOPs.
// ---------------------------------------------------------------------------
__global__ __launch_bounds__(128)
void mlp_kernel(const float* __restrict__ g,
                const float* __restrict__ Wl1, const float* __restrict__ bl1,
                const float* __restrict__ Wl2, const float* __restrict__ bl2,
                const float* __restrict__ Wl3, const float* __restrict__ bl3,
                float* __restrict__ out) {
    __shared__ float sg1[64 * 128];
    __shared__ float sg2[64 * 64];
    int tid = threadIdx.x;

    for (int idx = tid; idx < 64 * 128; idx += blockDim.x) {
        int row = idx >> 7, col = idx & 127;
        float s = bl1[col];
        #pragma unroll 4
        for (int k = 0; k < 64; ++k) s += g[row * 64 + k] * Wl1[k * 128 + col];
        sg1[idx] = s > 0.0f ? s : NEG_SLOPE * s;
    }
    __syncthreads();

    for (int idx = tid; idx < 64 * 64; idx += blockDim.x) {
        int row = idx >> 6, col = idx & 63;
        float s = bl2[col];
        #pragma unroll 4
        for (int k = 0; k < 128; ++k) s += sg1[row * 128 + k] * Wl2[k * 64 + col];
        sg2[idx] = s > 0.0f ? s : NEG_SLOPE * s;
    }
    __syncthreads();

    if (tid < 64) {
        float s = bl3[0];
        #pragma unroll 4
        for (int k = 0; k < 64; ++k) s += sg2[tid * 64 + k] * Wl3[k];
        out[tid] = s;
    }
}

// ---------------------------------------------------------------------------
// Launch
// ---------------------------------------------------------------------------
extern "C" void kernel_launch(void* const* d_in, const int* in_sizes, int n_in,
                              void* d_out, int out_size, void* d_ws, size_t ws_size,
                              hipStream_t stream) {
    const float* x     = (const float*)d_in[0];
    const int*   eidx  = (const int*)  d_in[1];
    const int*   batch = (const int*)  d_in[2];
    const float* W1    = (const float*)d_in[3];
    const float* b1    = (const float*)d_in[4];
    const float* W2    = (const float*)d_in[5];
    const float* b2    = (const float*)d_in[6];
    const float* Wl1   = (const float*)d_in[7];
    const float* bl1   = (const float*)d_in[8];
    const float* Wl2   = (const float*)d_in[9];
    const float* bl2   = (const float*)d_in[10];
    const float* Wl3   = (const float*)d_in[11];
    const float* bl3   = (const float*)d_in[12];
    float* out = (float*)d_out;

    const int N = in_sizes[0] / 128;   // 100000
    const int E = in_sizes[1] / 2;     // 1600000
    const int* src = eidx;
    const int* dst = eidx + E;

    // workspace layout (floats)
    float* ws   = (float*)d_ws;
    float* dis  = ws;                         // N
    float* h1   = dis  + N;                   // N*32
    float* agg1 = h1   + (size_t)N * 32;      // N*32
    float* h2   = agg1 + (size_t)N * 32;      // N*64
    float* agg2 = h2   + (size_t)N * 64;      // N*64
    float* g    = agg2 + (size_t)N * 64;      // 64*64

    // packed f16 weight fragments, 32B-aligned
    uintptr_t wp = (uintptr_t)(g + 64 * 64);
    wp = (wp + 31) & ~(uintptr_t)31;
    _Float16* Wf1 = (_Float16*)wp;            // 128*32 halves (8 KB)
    _Float16* Wf2 = Wf1 + 128 * 32;           // 32*64 halves (4 KB)

    const int T = 256;
    auto blk = [&](long long n) { return (unsigned)((n + T - 1) / T); };

    // 0. pack weights to f16 B-fragment layout (tiny, one-shot per call)
    pack_w_kernel<<<blk(128 * 32), T, 0, stream>>>(W1, Wf1, 32, 128 * 32);
    pack_w_kernel<<<blk(32 * 64), T, 0, stream>>>(W2, Wf2, 64, 32 * 64);

    // 1. degree (with self loop) -> dis = rsqrt(deg)
    fill_kernel<<<blk(N), T, 0, stream>>>(dis, 1.0f, N);
    deg_accum_kernel<<<blk(E), T, 0, stream>>>(dst, dis, E);
    rsqrt_kernel<<<blk(N), T, 0, stream>>>(dis, N);

    // 2. h1 = x @ W1   (WMMA, one wave per 16 rows)
    gemm_wmma_kernel<128, 32><<<N / 16, 32, 0, stream>>>(x, (const v16h*)Wf1, h1);

    // 3. layer-1 aggregation + bias + lrelu (in place in agg1)
    self_init_kernel<<<blk((long long)N * 32), T, 0, stream>>>(h1, dis, agg1, N, 32);
    edge_accum_kernel<32><<<blk((long long)E * 8), T, 0, stream>>>(h1, src, dst, dis, agg1, E);
    bias_act_kernel<<<blk((long long)N * 32), T, 0, stream>>>(agg1, b1, N, 32, 1);

    // 4. h2 = x2 @ W2  (WMMA)
    gemm_wmma_kernel<32, 64><<<N / 16, 32, 0, stream>>>(agg1, (const v16h*)Wf2, h2);

    // 5. layer-2 aggregation + bias (no activation)
    self_init_kernel<<<blk((long long)N * 64), T, 0, stream>>>(h2, dis, agg2, N, 64);
    edge_accum_kernel<64><<<blk((long long)E * 16), T, 0, stream>>>(h2, src, dst, dis, agg2, E);
    bias_act_kernel<<<blk((long long)N * 64), T, 0, stream>>>(agg2, b2, N, 64, 0);

    // 6. global max pool -> g [64,64]
    fill_kernel<<<blk(64 * 64), T, 0, stream>>>(g, -INFINITY, 64 * 64);
    pool_kernel<<<blk((long long)N * 64), T, 0, stream>>>(agg2, batch, g, N);

    // 7. tiny MLP head
    mlp_kernel<<<1, 128, 0, stream>>>(g, Wl1, bl1, Wl2, bl2, Wl3, bl3, out);
}
